// MLA_7894149890476
// MI455X (gfx1250) — matrix operand order repaired and checked by reading, not compiled, not run on hardware
//
#include <hip/hip_runtime.h>
#include <hip/hip_bf16.h>

// ---------------------------------------------------------------------------
// Types for gfx1250 WMMA (wave32): v_wmma_f32_16x16x32_bf16
// ---------------------------------------------------------------------------
typedef __attribute__((ext_vector_type(16))) __bf16 bf16x16;
typedef __attribute__((ext_vector_type(8)))  __bf16 bf16x8;
typedef __attribute__((ext_vector_type(8)))  float  f32x8;

static __device__ __forceinline__ f32x8 zero8() {
    f32x8 z;
#pragma unroll
    for (int i = 0; i < 8; ++i) z[i] = 0.0f;
    return z;
}

static __device__ __forceinline__ f32x8 wmma_bf16(bf16x16 a, bf16x16 b, f32x8 c) {
    return __builtin_amdgcn_wmma_f32_16x16x32_bf16(
        false, a, false, b, (short)0, c, false, false);
}

// Build a 16-element fragment from two contiguous 8-element (16B) chunks.
static __device__ __forceinline__ bf16x16 make_frag(const __bf16* lo_p, const __bf16* hi_p) {
    bf16x8 lo = *(const bf16x8*)lo_p;
    bf16x8 hi = *(const bf16x8*)hi_p;
    bf16x16 r;
#pragma unroll
    for (int j = 0; j < 8; ++j) { r[j] = lo[j]; r[j + 8] = hi[j]; }
    return r;
}

union U8 { uint4 u; __bf16 h[8]; };

// Pack two f32 -> packed bf16 pair (lowers to v_cvt_pk_bf16_f32).
static __device__ __forceinline__ unsigned pack_bf16(float a, float b) {
    union { __bf16 h[2]; unsigned u; } x;
    x.h[0] = (__bf16)a; x.h[1] = (__bf16)b;
    return x.u;
}
static __device__ __forceinline__ unsigned pack2h(__bf16 a, __bf16 b) {
    union { __bf16 h[2]; unsigned u; } x;
    x.h[0] = a; x.h[1] = b;
    return x.u;
}

// ---------------------------------------------------------------------------
// Generic GEMM: C[M,N] = A[M,K] * B[K,N]   (A * B^T when transB!=0)
// f32 in/out, bf16 WMMA compute, f32 accumulate.
// 256 threads / 8 waves; tile 128x128, K-step 32; double-buffered LDS;
// branchless clamped loads; K-paired packed ds_store_b32; prefetch kt+2.
// ---------------------------------------------------------------------------
#define GBM 128
#define GBN 128
#define GBK 32
#define LDPAD 40   // 80B row stride: 16B multiple, bank-conflict free

__global__ __launch_bounds__(256) void gemm_bf16_wmma(
    const float* __restrict__ A, const float* __restrict__ B, float* __restrict__ C,
    int M, int N, int K, int lda, int ldb, int ldc, int transB)
{
    __shared__ __bf16 Alds[2][GBM][LDPAD];   // [buf][m][k]
    __shared__ __bf16 Blds[2][GBN][LDPAD];   // [buf][n][k]

    const int tid  = threadIdx.x;
    const int lane = tid & 31;
    const int wave = tid >> 5;
    const int wm   = wave >> 2;          // 0..1
    const int wn   = wave & 3;           // 0..3
    const int half = lane >> 4;          // 0/1
    const int l16  = lane & 15;
    const int m0   = blockIdx.y * GBM;
    const int n0   = blockIdx.x * GBN;

    f32x8 acc[4][2];
#pragma unroll
    for (int mi = 0; mi < 4; ++mi)
#pragma unroll
        for (int ni = 0; ni < 2; ++ni) acc[mi][ni] = zero8();

    const int ksteps = (K + GBK - 1) / GBK;

    float2 av[8], bv[8];

    // Branchless clamped tile load into registers; each thread owns 8 adjacent
    // K-pairs so converts fuse into v_cvt_pk_bf16_f32 + packed b32 LDS stores.
    auto load_tile = [&](int kt) {
        const int k0 = kt * GBK;
#pragma unroll
        for (int i = 0; i < 8; ++i) {
            int p = tid + (i << 8);          // pair index 0..2047
            int r = p >> 4;                  // row 0..127
            int c = (p & 15) << 1;           // even k 0..30
            int gm = m0 + r;
            bool okm = gm < M;
            gm = okm ? gm : M - 1;
            int gk0 = k0 + c, gk1 = gk0 + 1;
            bool ok0 = okm && (gk0 < K), ok1 = okm && (gk1 < K);
            gk0 = gk0 < K ? gk0 : K - 1;
            gk1 = gk1 < K ? gk1 : K - 1;
            const float* ap = A + (size_t)gm * lda;
            float x = ap[gk0], y = ap[gk1];
            av[i].x = ok0 ? x : 0.0f;
            av[i].y = ok1 ? y : 0.0f;
        }
        if (!transB) {
#pragma unroll
            for (int i = 0; i < 8; ++i) {
                int p = tid + (i << 8);
                int nn = p & 127;            // n within tile (coalesced)
                int c  = (p >> 7) << 1;      // even k
                int gn = n0 + nn;
                bool okn = gn < N;
                gn = okn ? gn : N - 1;
                int gk0 = (kt * GBK) + c, gk1 = gk0 + 1;
                bool ok0 = okn && (gk0 < K), ok1 = okn && (gk1 < K);
                gk0 = gk0 < K ? gk0 : K - 1;
                gk1 = gk1 < K ? gk1 : K - 1;
                float x = B[(size_t)gk0 * ldb + gn];
                float y = B[(size_t)gk1 * ldb + gn];
                bv[i].x = ok0 ? x : 0.0f;
                bv[i].y = ok1 ? y : 0.0f;
            }
        } else {
#pragma unroll
            for (int i = 0; i < 8; ++i) {
                int p = tid + (i << 8);
                int nn = p >> 4;
                int c  = (p & 15) << 1;
                int gn = n0 + nn;
                bool okn = gn < N;
                gn = okn ? gn : N - 1;
                int gk0 = (kt * GBK) + c, gk1 = gk0 + 1;
                bool ok0 = okn && (gk0 < K), ok1 = okn && (gk1 < K);
                gk0 = gk0 < K ? gk0 : K - 1;
                gk1 = gk1 < K ? gk1 : K - 1;
                const float* bp = B + (size_t)gn * ldb;
                float x = bp[gk0], y = bp[gk1];
                bv[i].x = ok0 ? x : 0.0f;
                bv[i].y = ok1 ? y : 0.0f;
            }
        }
    };

    auto store_tile = [&](int buf) {
#pragma unroll
        for (int i = 0; i < 8; ++i) {
            int p = tid + (i << 8);
            int r = p >> 4, c = (p & 15) << 1;
            *(unsigned*)&Alds[buf][r][c] = pack_bf16(av[i].x, av[i].y);
        }
        if (!transB) {
#pragma unroll
            for (int i = 0; i < 8; ++i) {
                int p = tid + (i << 8);
                int nn = p & 127, c = (p >> 7) << 1;
                *(unsigned*)&Blds[buf][nn][c] = pack_bf16(bv[i].x, bv[i].y);
            }
        } else {
#pragma unroll
            for (int i = 0; i < 8; ++i) {
                int p = tid + (i << 8);
                int nn = p >> 4, c = (p & 15) << 1;
                *(unsigned*)&Blds[buf][nn][c] = pack_bf16(bv[i].x, bv[i].y);
            }
        }
    };

    load_tile(0);
    store_tile(0);

    for (int kt = 0; kt < ksteps; ++kt) {
        __syncthreads();
        const int cur = kt & 1;
        const bool more = (kt + 1) < ksteps;
        if (more) load_tile(kt + 1);          // overlap with WMMA below

        // Ride ahead of the double buffer: prefetch tile kt+2 (speculative,
        // lowers to global_prefetch_b8; OOB translations silently dropped).
        if (kt + 2 < ksteps) {
            const int kp2 = (kt + 2) * GBK;
            __builtin_prefetch(A + (size_t)(m0 + (tid >> 1)) * lda + kp2, 0, 0);
            if (!transB)
                __builtin_prefetch(B + (size_t)(kp2 + (tid & 31)) * ldb + n0, 0, 0);
            else
                __builtin_prefetch(B + (size_t)(n0 + (tid & 127)) * ldb + kp2, 0, 0);
        }

        bf16x16 bfr[2];
#pragma unroll
        for (int ni = 0; ni < 2; ++ni) {
            const __bf16* p = &Blds[cur][wn * 32 + ni * 16 + l16][half * 16];
            bfr[ni] = make_frag(p, p + 8);
        }
#pragma unroll
        for (int mi = 0; mi < 4; ++mi) {
            const __bf16* p = &Alds[cur][wm * 64 + mi * 16 + l16][half * 16];
            bf16x16 af = make_frag(p, p + 8);
            acc[mi][0] = wmma_bf16(af, bfr[0], acc[mi][0]);
            acc[mi][1] = wmma_bf16(af, bfr[1], acc[mi][1]);
        }

        if (more) store_tile(cur ^ 1);
    }

    // C layout: VGPR v, lane l -> (M = v + 8*(l>=16), N = l%16)
#pragma unroll
    for (int mi = 0; mi < 4; ++mi)
#pragma unroll
        for (int ni = 0; ni < 2; ++ni)
#pragma unroll
            for (int v = 0; v < 8; ++v) {
                int m = m0 + wm * 64 + mi * 16 + v + 8 * half;
                int n = n0 + wn * 32 + ni * 16 + l16;
                if (m < M && n < N) C[(size_t)m * ldc + n] = acc[mi][ni][v];
            }
}

// ---------------------------------------------------------------------------
// Row LayerNorm: one block per row.
// ---------------------------------------------------------------------------
__global__ __launch_bounds__(256) void layernorm_rows(
    const float* __restrict__ in, float* __restrict__ out,
    const float* __restrict__ w, const float* __restrict__ b,
    int n, int ld_in, int ld_out)
{
    const int row = blockIdx.x;
    const float* x = in + (size_t)row * ld_in;
    float* y = out + (size_t)row * ld_out;
    __shared__ float red[2][8];

    float s = 0.0f, ss = 0.0f;
    for (int i = threadIdx.x; i < n; i += 256) {
        float v = x[i]; s += v; ss += v * v;
    }
#pragma unroll
    for (int m = 1; m < 32; m <<= 1) {
        s  += __shfl_xor(s,  m, 32);
        ss += __shfl_xor(ss, m, 32);
    }
    const int wave = threadIdx.x >> 5, lane = threadIdx.x & 31;
    if (lane == 0) { red[0][wave] = s; red[1][wave] = ss; }
    __syncthreads();
    float ts = 0.0f, tss = 0.0f;
#pragma unroll
    for (int i = 0; i < 8; ++i) { ts += red[0][i]; tss += red[1][i]; }
    const float mean = ts / (float)n;
    const float var  = tss / (float)n - mean * mean;
    const float rstd = rsqrtf(var + 1e-5f);
    for (int i = threadIdx.x; i < n; i += 256)
        y[i] = (x[i] - mean) * rstd * w[i] + b[i];
}

// ---------------------------------------------------------------------------
// RoPE + repack to bf16 (B,H,S,128). Q is pre-scaled by 1/sqrt(128).
// ---------------------------------------------------------------------------
static __device__ __forceinline__ float rope_val(const float* v64, int i, int s) {
    float x   = v64[i];
    float rot = (i < 32) ? -v64[i + 32] : v64[i - 32];
    float f   = __powf(10000.0f, -(float)(i & 31) / 64.0f);
    float sv, cv;
    __sincosf((float)s * f, &sv, &cv);
    return x * cv + rot * sv;
}

__global__ __launch_bounds__(256) void build_q_kernel(
    const float* __restrict__ Q, __bf16* __restrict__ qb, int S, int total)
{
    int idx = blockIdx.x * 256 + threadIdx.x;
    if (idx >= total) return;
    int d = idx & 127;
    int t = idx >> 7;
    int h = t & 15; t >>= 4;
    int s = t % S;  int b = t / S;
    const float* src = Q + (((size_t)b * S + s) * 2048 + (size_t)h * 128);
    float val = (d < 64) ? src[d] : rope_val(src + 64, d - 64, s);
    val *= 0.08838834764831845f;    // fold 1/sqrt(DH) into Q
    qb[(((size_t)b * 16 + h) * S + s) * 128 + d] = (__bf16)val;
}

__global__ __launch_bounds__(256) void build_kv_kernel(
    const float* __restrict__ KV, const float* __restrict__ ckv,
    __bf16* __restrict__ Kb, __bf16* __restrict__ Vb, int S, int total)
{
    int idx = blockIdx.x * 256 + threadIdx.x;
    if (idx >= total) return;
    int d = idx & 127;
    int t = idx >> 7;
    int h = t & 15; t >>= 4;
    int s = t % S;  int b = t / S;
    const float* kvrow = KV + (((size_t)b * S + s) * 3072 + (size_t)h * 192);
    float kval;
    if (d < 64) {
        kval = kvrow[d];
    } else {
        const float* kr = ckv + ((size_t)b * S + s) * 1429 + 1365;
        kval = rope_val(kr, d - 64, s);
    }
    size_t o = (((size_t)b * 16 + h) * S + s) * 128 + d;
    Kb[o] = (__bf16)kval;
    Vb[o] = (__bf16)kvrow[64 + d];
}

// ---------------------------------------------------------------------------
// Flash attention (causal): 128 q-rows/block, 8 waves x 16 rows.
// Double-buffered K/V tiles; uint4 global traffic; packed transpose stores.
// ---------------------------------------------------------------------------
__global__ __launch_bounds__(256) void flash_attn_kernel(
    const __bf16* __restrict__ Qb, const __bf16* __restrict__ Kb,
    const __bf16* __restrict__ Vb, float* __restrict__ Out, int S)
{
    __shared__ __bf16 Klds[2][32][136];    // [buf][t][d], 272B stride
    __shared__ __bf16 Vlds[2][128][40];    // [buf][d][t], 80B stride
    __shared__ __bf16 Plds[8][16][40];     // per-wave P tile

    const int b = blockIdx.z, h = blockIdx.y;
    const int q0 = blockIdx.x * 128;
    const int tid = threadIdx.x, lane = tid & 31, wave = tid >> 5;
    const int half = lane >> 4, l16 = lane & 15;

    const __bf16* qbase = Qb + (((size_t)b * 16 + h) * S) * 128;
    const __bf16* kbase = Kb + (((size_t)b * 16 + h) * S) * 128;
    const __bf16* vbase = Vb + (((size_t)b * 16 + h) * S) * 128;

    // Q fragments (rows q0 + wave*16 + l16), already scaled by 1/sqrt(DH)
    bf16x16 aq[4];
    {
        const __bf16* qrow = qbase + (size_t)(q0 + wave * 16 + l16) * 128;
#pragma unroll
        for (int c = 0; c < 4; ++c) {
            const __bf16* p = qrow + c * 32 + half * 16;
            aq[c] = make_frag(p, p + 8);
        }
    }

    f32x8 oacc[8];
#pragma unroll
    for (int c = 0; c < 8; ++c) oacc[c] = zero8();
    float rmax[8], rsum[8];
#pragma unroll
    for (int v = 0; v < 8; ++v) { rmax[v] = -3.0e38f; rsum[v] = 0.0f; }

    // Each thread owns key-row pair (t2, t2+1) at one 8-elem d-chunk:
    // 4x global b128 in, K stored wide, V transposed via packed b32 pairs.
    const int t2 = (tid >> 4) << 1;      // even key row 0..30
    const int dc = tid & 15;             // d-chunk
    uint4 kr0, kr1, vr0, vr1;
    auto load_kv = [&](int t0) {
        const __bf16* kp = kbase + (size_t)(t0 + t2) * 128 + dc * 8;
        const __bf16* vp = vbase + (size_t)(t0 + t2) * 128 + dc * 8;
        kr0 = *(const uint4*)kp;
        kr1 = *(const uint4*)(kp + 128);
        vr0 = *(const uint4*)vp;
        vr1 = *(const uint4*)(vp + 128);
    };
    auto store_kv = [&](int buf) {
        *(uint4*)&Klds[buf][t2][dc * 8]     = kr0;   // ds_store_b128
        *(uint4*)&Klds[buf][t2 + 1][dc * 8] = kr1;
        U8 a, c; a.u = vr0; c.u = vr1;
#pragma unroll
        for (int j = 0; j < 8; ++j)
            *(unsigned*)&Vlds[buf][dc * 8 + j][t2] = pack2h(a.h[j], c.h[j]);
    };

    const int nt = (q0 + 128) >> 5;   // causal tile count
    load_kv(0);
    store_kv(0);

    for (int it = 0; it < nt; ++it) {
        __syncthreads();
        const int cur = it & 1;
        const int t0 = it << 5;
        const bool more = (it + 1) < nt;
        if (more) load_kv(t0 + 32);       // overlap with compute

        // Scores: two 16x16 key tiles, K-chained over d (4 x WMMA each)
        f32x8 s0 = zero8(), s1 = zero8();
#pragma unroll
        for (int c = 0; c < 4; ++c) {
            const __bf16* p0 = &Klds[cur][l16][c * 32 + half * 16];
            s0 = wmma_bf16(aq[c], make_frag(p0, p0 + 8), s0);
            const __bf16* p1 = &Klds[cur][16 + l16][c * 32 + half * 16];
            s1 = wmma_bf16(aq[c], make_frag(p1, p1 + 8), s1);
        }

        // Online softmax per row (row = q0 + wave*16 + v + 8*half)
        const int qrb = q0 + wave * 16;
#pragma unroll
        for (int v = 0; v < 8; ++v) {
            const int qr = qrb + v + 8 * half;
            float a0 = s0[v]; if (t0 + l16 > qr)      a0 = -3.0e38f;
            float a1 = s1[v]; if (t0 + 16 + l16 > qr) a1 = -3.0e38f;
            float m = fmaxf(a0, a1);
#pragma unroll
            for (int msk = 1; msk < 16; msk <<= 1) m = fmaxf(m, __shfl_xor(m, msk, 32));
            const float nmax = fmaxf(rmax[v], m);
            const float corr = __expf(rmax[v] - nmax);
            rmax[v] = nmax;
            const float p0v = __expf(a0 - nmax);
            const float p1v = __expf(a1 - nmax);
            s0[v] = p0v; s1[v] = p1v;
            float ts = p0v + p1v;
#pragma unroll
            for (int msk = 1; msk < 16; msk <<= 1) ts += __shfl_xor(ts, msk, 32);
            rsum[v] = rsum[v] * corr + ts;
#pragma unroll
            for (int c = 0; c < 8; ++c) oacc[c][v] *= corr;
        }

        // P (C layout) -> per-wave LDS tile -> A-fragment layout
#pragma unroll
        for (int v = 0; v < 8; ++v) {
            Plds[wave][v + 8 * half][l16]      = (__bf16)s0[v];
            Plds[wave][v + 8 * half][16 + l16] = (__bf16)s1[v];
        }
        asm volatile("s_wait_dscnt 0x0" ::: "memory");   // same-wave LDS RAW

        bf16x16 ap;
        {
            const __bf16* p = &Plds[wave][l16][half * 16];
            ap = make_frag(p, p + 8);
        }
#pragma unroll
        for (int c = 0; c < 8; ++c) {
            const __bf16* p = &Vlds[cur][c * 16 + l16][half * 16];
            oacc[c] = wmma_bf16(ap, make_frag(p, p + 8), oacc[c]);
        }

        if (more) store_kv(cur ^ 1);
    }

    // Normalize and store: Out[b][s][h*128 + d] (head-major columns)
    const int qrb = q0 + wave * 16;
#pragma unroll
    for (int v = 0; v < 8; ++v) {
        const float inv = 1.0f / rsum[v];
        const int m = qrb + v + 8 * half;
#pragma unroll
        for (int c = 0; c < 8; ++c) {
            const int d = c * 16 + l16;
            Out[((size_t)b * S + m) * 2048 + (size_t)h * 128 + d] = oacc[c][v] * inv;
        }
    }
}

// ---------------------------------------------------------------------------
// Host launcher
// ---------------------------------------------------------------------------
extern "C" void kernel_launch(void* const* d_in, const int* in_sizes, int n_in,
                              void* d_out, int out_size, void* d_ws, size_t ws_size,
                              hipStream_t stream) {
    (void)in_sizes; (void)n_in; (void)out_size; (void)ws_size;

    const float* x       = (const float*)d_in[0];
    const float* W_dq    = (const float*)d_in[1];
    const float* W_uq    = (const float*)d_in[2];
    const float* q_ln_w  = (const float*)d_in[3];
    const float* q_ln_b  = (const float*)d_in[4];
    const float* W_dkv   = (const float*)d_in[5];
    const float* W_ukv   = (const float*)d_in[6];
    const float* kv_ln_w = (const float*)d_in[7];
    const float* kv_ln_b = (const float*)d_in[8];
    const float* W_o     = (const float*)d_in[9];

    const int B = 2, S = 2048, D = 2048, H = 16;
    const int QP = 1024, KVP = 1365, CKV = 1429, KVN = 3072;
    const int BS = B * S;

    float* out = (float*)d_out;
    float* ckv = out + (size_t)BS * D;

    char* ws = (char*)d_ws;
    float* cq   = (float*)ws;  ws += (size_t)BS * QP  * 4;
    float* Qf   = (float*)ws;  ws += (size_t)BS * D   * 4;
    float* kvl  = (float*)ws;  ws += (size_t)BS * KVP * 4;
    float* KVf  = (float*)ws;  ws += (size_t)BS * KVN * 4;
    __bf16* qb  = (__bf16*)ws; ws += (size_t)BS * D * 2;
    __bf16* Kb  = (__bf16*)ws; ws += (size_t)BS * D * 2;
    __bf16* Vb  = (__bf16*)ws; ws += (size_t)BS * D * 2;
    float* attn = (float*)ws;  ws += (size_t)BS * D * 4;

    const dim3 blk(256);
    auto gg = [](int M, int N) { return dim3((N + GBN - 1) / GBN, (M + GBM - 1) / GBM); };

    gemm_bf16_wmma<<<gg(BS, QP),  blk, 0, stream>>>(x, W_dq,  cq,  BS, QP,  D, D, QP,  QP,  0);
    gemm_bf16_wmma<<<gg(BS, CKV), blk, 0, stream>>>(x, W_dkv, ckv, BS, CKV, D, D, CKV, CKV, 0);
    layernorm_rows<<<BS, blk, 0, stream>>>(cq,  cq,  q_ln_w,  q_ln_b,  QP,  QP,  QP);
    layernorm_rows<<<BS, blk, 0, stream>>>(ckv, kvl, kv_ln_w, kv_ln_b, KVP, CKV, KVP);
    gemm_bf16_wmma<<<gg(BS, D),   blk, 0, stream>>>(cq,  W_uq,  Qf,  BS, D,   QP,  QP,  D,   D,   0);
    gemm_bf16_wmma<<<gg(BS, KVN), blk, 0, stream>>>(kvl, W_ukv, KVf, BS, KVN, KVP, KVP, KVN, KVN, 0);
    const int total = B * S * H * 128;
    build_q_kernel <<<(total + 255) / 256, blk, 0, stream>>>(Qf, qb, S, total);
    build_kv_kernel<<<(total + 255) / 256, blk, 0, stream>>>(KVf, ckv, Kb, Vb, S, total);
    flash_attn_kernel<<<dim3(S / 128, H, B), blk, 0, stream>>>(qb, Kb, Vb, attn, S);
    gemm_bf16_wmma<<<gg(BS, D), blk, 0, stream>>>(attn, W_o, out, BS, D, D, D, D, D, 1);
}
